// DeepseekV4Attention_67637144978440
// MI455X (gfx1250) — 16-bit and 8-bit matrix operand order repaired and checked by reading, compiled, not run on hardware
//
#include <hip/hip_runtime.h>

#define BATCH 32
#define HEADS 64
#define DIM   512
#define ROPE_D 64
#define HALF_R 32
#define NBLK  2048
#define BS    64
#define MBPB  64
#define SMAX  4096
#define IH    64
#define ID    128
#define TOPK  512
#define SCALE_F 0.044194173824159216f

#define KEEP_PAD 528
#define NPAD 544

typedef _Float16 f16;
typedef __attribute__((ext_vector_type(16))) _Float16 v16h;
typedef __attribute__((ext_vector_type(8)))  float    v8f;
typedef __attribute__((ext_vector_type(8)))  int      v8i;

#define OFF_QQ   0u
#define OFF_QS   262144u
#define OFF_QR   270336u
#define OFF_ISC  2367488u
#define OFF_KIDX 2891776u
#define OFF_KCNT 2959360u

__device__ __forceinline__ int kpairA16(int v, int hi) {
    return (v < 4 ? 2 * v : 16 + 2 * (v - 4)) + 8 * hi;
}
__device__ __forceinline__ int kpairB16(int v, int hi) {
    return (v < 4 ? 2 * v : 16 + 2 * (v - 4)) + 8 * hi;
}
__device__ __forceinline__ int kbaseA8(int v, int hi) {
    return (v >> 2) * 32 + ((v >> 1) & 1) * 16 + (v & 1) * 4 + 8 * hi;
}
__device__ __forceinline__ int kbaseB8(int v, int hi) {
    return (v >> 2) * 32 + ((v >> 1) & 1) * 16 + (v & 1) * 4 + 8 * hi;
}

__device__ __forceinline__ unsigned pack_h2(float a, float b) {
    union { f16 h[2]; unsigned u; } x;
    x.h[0] = (f16)a; x.h[1] = (f16)b;
    return x.u;
}

__global__ __launch_bounds__(128) void k_quant_q(const float* __restrict__ qidx,
                                                 unsigned* __restrict__ qq,
                                                 float* __restrict__ qs) {
    const int b = blockIdx.x;
    const int w = threadIdx.x >> 5, lane = threadIdx.x & 31;
    for (int i = 0; i < 16; ++i) {
        const int h = w * 16 + i;
        const float4 x = ((const float4*)(qidx + ((size_t)b * IH + h) * ID))[lane];
        float am = fmaxf(fmaxf(fabsf(x.x), fabsf(x.y)), fmaxf(fabsf(x.z), fabsf(x.w)));
        #pragma unroll
        for (int off = 16; off; off >>= 1) am = fmaxf(am, __shfl_xor(am, off, 32));
        am = fmaxf(am, 1e-12f);
        float sc = am / 127.0f;
        if (sc < 1e-6f) sc = 1.0f;
        const float inv = 1.0f / sc;
        int i0 = (int)fminf(fmaxf(rintf(x.x * inv), -128.f), 127.f);
        int i1 = (int)fminf(fmaxf(rintf(x.y * inv), -128.f), 127.f);
        int i2 = (int)fminf(fmaxf(rintf(x.z * inv), -128.f), 127.f);
        int i3 = (int)fminf(fmaxf(rintf(x.w * inv), -128.f), 127.f);
        qq[((size_t)b * IH + h) * (ID / 4) + lane] =
            (unsigned)(i0 & 255) | ((unsigned)(i1 & 255) << 8) |
            ((unsigned)(i2 & 255) << 16) | ((unsigned)(i3 & 255) << 24);
        if (lane == 0) qs[b * IH + h] = sc;
    }
}

__global__ __launch_bounds__(64) void k_rope(const float* __restrict__ q,
                                             const float* __restrict__ cosb,
                                             const float* __restrict__ sinb,
                                             f16* __restrict__ qr) {
    const int h = blockIdx.x, b = blockIdx.y;
    const float* src = q + ((size_t)b * HEADS + h) * DIM;
    f16* dst = qr + ((size_t)b * HEADS + h) * DIM;
    for (int d = threadIdx.x; d < DIM - ROPE_D; d += 64) dst[d] = (f16)src[d];
    if (threadIdx.x < HALF_R) {
        const int j = threadIdx.x;
        const float c = cosb[b * HALF_R + j], s = sinb[b * HALF_R + j];
        const float e = src[DIM - ROPE_D + 2 * j], o = src[DIM - ROPE_D + 2 * j + 1];
        dst[DIM - ROPE_D + 2 * j]     = (f16)(e * c - o * s);
        dst[DIM - ROPE_D + 2 * j + 1] = (f16)(e * s + o * c);
    }
}

__global__ __launch_bounds__(128) void k_iscore(const float* __restrict__ kidx_cache,
                                                const int* __restrict__ btab,
                                                const int* __restrict__ seq_lens,
                                                const unsigned* __restrict__ qq,
                                                const float* __restrict__ qs,
                                                float* __restrict__ iscore) {
    __shared__ unsigned kq[BS * (ID / 4)];
    __shared__ float kss[BS];
    __shared__ float qsl[IH];

    const int blk = blockIdx.x, b = blockIdx.y;
    const int tid = threadIdx.x, w = tid >> 5, lane = tid & 31;
    const int bt = btab[b * MBPB + blk];
    const float* kb = kidx_cache + (size_t)bt * BS * ID;

    for (int i = 0; i < 16; ++i) {
        const int r = w * 16 + i;
        const float4 x = ((const float4*)(kb + (size_t)r * ID))[lane];
        float am = fmaxf(fmaxf(fabsf(x.x), fabsf(x.y)), fmaxf(fabsf(x.z), fabsf(x.w)));
        #pragma unroll
        for (int off = 16; off; off >>= 1) am = fmaxf(am, __shfl_xor(am, off, 32));
        am = fmaxf(am, 1e-12f);
        float sc = am / 127.0f;
        if (sc < 1e-6f) sc = 1.0f;
        const float inv = 1.0f / sc;
        int i0 = (int)fminf(fmaxf(rintf(x.x * inv), -128.f), 127.f);
        int i1 = (int)fminf(fmaxf(rintf(x.y * inv), -128.f), 127.f);
        int i2 = (int)fminf(fmaxf(rintf(x.z * inv), -128.f), 127.f);
        int i3 = (int)fminf(fmaxf(rintf(x.w * inv), -128.f), 127.f);
        kq[r * (ID / 4) + lane] =
            (unsigned)(i0 & 255) | ((unsigned)(i1 & 255) << 8) |
            ((unsigned)(i2 & 255) << 16) | ((unsigned)(i3 & 255) << 24);
        if (lane == 0) kss[r] = sc;
    }
    if (tid < IH) qsl[tid] = qs[b * IH + tid];
    __syncthreads();

    const int nl = lane & 15, hi = lane >> 4;
    const unsigned* qqb = qq + (size_t)b * IH * (ID / 4);
    float tot = 0.f;

    for (int mt = 0; mt < 4; ++mt) {
        union { v8i v; int i[8]; unsigned u[8]; } A, B, C;
        #pragma unroll
        for (int r = 0; r < 8; ++r) C.i[r] = 0;
        #pragma unroll
        for (int kh = 0; kh < 2; ++kh) {
            const int head = mt * 16 + nl;
            const int row = w * 16 + nl;
            #pragma unroll
            for (int v = 0; v < 8; ++v)
                A.u[v] = qqb[head * (ID / 4) + ((kh * 64 + kbaseA8(v, hi)) >> 2)];
            #pragma unroll
            for (int v = 0; v < 8; ++v)
                B.u[v] = kq[row * (ID / 4) + ((kh * 64 + kbaseB8(v, hi)) >> 2)];
            C.v = __builtin_amdgcn_wmma_i32_16x16x64_iu8(true, A.v, true, B.v, C.v, false, false);
        }
        float p = 0.f;
        #pragma unroll
        for (int r = 0; r < 8; ++r) p += (float)C.i[r] * qsl[mt * 16 + 8 * hi + r];
        tot += p;
    }
    tot += __shfl_xor(tot, 16, 32);

    if (lane < 16) {
        const int r = w * 16 + lane;
        const int s = blk * BS + r;
        const int seql = seq_lens[b];
        iscore[(size_t)b * SMAX + s] = (s < seql) ? tot * kss[r] : -__builtin_inff();
    }
}

__global__ __launch_bounds__(256) void k_topk(const float* __restrict__ iscore,
                                              int* __restrict__ keep_idx,
                                              int* __restrict__ keep_cnt) {
    __shared__ unsigned keys[SMAX];
    __shared__ unsigned hist[256];
    __shared__ int eqc[256];
    __shared__ unsigned sPrefix;
    __shared__ int sKrem, sOut;

    const int b = blockIdx.x, tid = threadIdx.x;
    for (int s = tid; s < SMAX; s += 256) {
        const unsigned u = __float_as_uint(iscore[(size_t)b * SMAX + s]);
        keys[s] = (u & 0x80000000u) ? ~u : (u | 0x80000000u);
    }
    if (tid == 0) { sPrefix = 0; sKrem = TOPK; sOut = 0; }
    __syncthreads();

    for (int pass = 0; pass < 4; ++pass) {
        hist[tid] = 0;
        __syncthreads();
        const int shift = 24 - 8 * pass;
        const unsigned pref = sPrefix;
        for (int s = tid; s < SMAX; s += 256) {
            const unsigned k = keys[s];
            const bool part = (pass == 0) || ((k >> (shift + 8)) == pref);
            if (part) atomicAdd(&hist[(k >> shift) & 255u], 1u);
        }
        __syncthreads();
        if (tid == 0) {
            unsigned c = 0;
            const int krem = sKrem;
            for (int j = 255; j >= 0; --j) {
                c += hist[j];
                if ((int)c >= krem) {
                    sKrem = krem - (int)(c - hist[j]);
                    sPrefix = (sPrefix << 8) | (unsigned)j;
                    break;
                }
            }
        }
        __syncthreads();
    }
    const unsigned T = sPrefix;
    const int krem = sKrem;

    const int base = tid * 16;
    int ec = 0;
    for (int i = 0; i < 16; ++i) ec += (keys[base + i] == T) ? 1 : 0;
    eqc[tid] = ec;
    __syncthreads();
    if (tid == 0) {
        int ex = 0;
        for (int t = 0; t < 256; ++t) { const int tmp = eqc[t]; eqc[t] = ex; ex += tmp; }
    }
    __syncthreads();

    int running = eqc[tid];
    for (int i = 0; i < 16; ++i) {
        const int s = base + i;
        const unsigned k = keys[s];
        bool take = false;
        if (k > T) take = true;
        else if (k == T) { if (running < krem) take = true; ++running; }
        if (take) {
            const int slot = atomicAdd(&sOut, 1);
            keep_idx[b * KEEP_PAD + slot] = s;
        }
    }
    __syncthreads();
    if (tid == 0) {
        int tot = sOut;
        if (keys[0] < T) { keep_idx[b * KEEP_PAD + tot] = 0; ++tot; }
        keep_cnt[b] = tot;
    }
}

__global__ __launch_bounds__(256) void k_attn(const f16* __restrict__ qr,
                                              const float* __restrict__ kv,
                                              const int* __restrict__ btab,
                                              const int* __restrict__ keep_idx,
                                              const int* __restrict__ keep_cnt,
                                              const float* __restrict__ sink,
                                              float* __restrict__ out) {
    __shared__ __align__(16) char smem[52224];
    __shared__ float rcpsum[16];
    f16*   wexp   = (f16*)smem;
    float* scores = (float*)(smem + 17408);
    f16*   kT     = (f16*)(smem + 17408);

    const int ht = blockIdx.x, b = blockIdx.y;
    const int tid = threadIdx.x, w = tid >> 5, lane = tid & 31;
    const int nl = lane & 15, hi = lane >> 4;
    const int cnt = keep_cnt[b];
    const int* kidx = keep_idx + b * KEEP_PAD;
    const int* bt = btab + b * MBPB;

    float sinkr[8];
    #pragma unroll
    for (int r = 0; r < 8; ++r) sinkr[r] = sink[ht * 16 + 8 * hi + r];

    const unsigned* qru =
        (const unsigned*)(qr + ((size_t)b * HEADS + ht * 16 + nl) * DIM);
    for (int t = w; t < NPAD / 16; t += 8) {
        const int n = t * 16 + nl;
        const int pos = (n < cnt) ? kidx[n] : 0;
        const float* krow = kv + ((size_t)bt[pos >> 6] * BS + (pos & 63)) * DIM;
        union { v16h v; unsigned u[8]; } A, B;
        union { v8f v; float f[8]; } C;
        #pragma unroll
        for (int r = 0; r < 8; ++r) C.f[r] = 0.f;
        for (int kc = 0; kc < DIM / 32; ++kc) {
            #pragma unroll
            for (int v = 0; v < 8; ++v)
                A.u[v] = qru[(kc * 32 + kpairA16(v, hi)) >> 1];
            #pragma unroll
            for (int v = 0; v < 8; ++v) {
                const float2 f2 = *(const float2*)(krow + kc * 32 + kpairB16(v, hi));
                B.u[v] = pack_h2(f2.x, f2.y);
            }
            C.v = __builtin_amdgcn_wmma_f32_16x16x32_f16(false, A.v, false, B.v,
                                                         (short)0, C.v, false, false);
        }
        #pragma unroll
        for (int r = 0; r < 8; ++r) {
            const int m = 8 * hi + r;
            float val = -__builtin_inff();
            if (n < cnt) val = C.f[r] * SCALE_F + (pos == 0 ? sinkr[r] : 0.f);
            scores[m * NPAD + t * 16 + nl] = val;
        }
    }
    __syncthreads();

    for (int hh = 0; hh < 2; ++hh) {
        const int m = w * 2 + hh;
        float mx = -__builtin_inff();
        for (int n = lane; n < NPAD; n += 32) mx = fmaxf(mx, scores[m * NPAD + n]);
        #pragma unroll
        for (int off = 16; off; off >>= 1) mx = fmaxf(mx, __shfl_xor(mx, off, 32));
        float sm = 0.f;
        for (int n = lane; n < NPAD; n += 32) {
            const float e = expf(scores[m * NPAD + n] - mx);
            wexp[m * NPAD + n] = (f16)e;
            sm += e;
        }
        #pragma unroll
        for (int off = 16; off; off >>= 1) sm += __shfl_xor(sm, off, 32);
        if (lane == 0) rcpsum[m] = 1.0f / sm;
    }
    __syncthreads();

    union { v8f v; float f[8]; } acc[4];
    #pragma unroll
    for (int t = 0; t < 4; ++t)
        #pragma unroll
        for (int r = 0; r < 8; ++r) acc[t].f[r] = 0.f;

    const unsigned* wexpu = (const unsigned*)wexp;
    for (int kc = 0; kc < NPAD / 32; ++kc) {
        __syncthreads();
        {
            const int p = tid >> 3;
            const int n = kc * 32 + p;
            const bool ok = (n < cnt);
            const int pos = ok ? kidx[n] : 0;
            const float* krow = kv + ((size_t)bt[pos >> 6] * BS + (pos & 63)) * DIM;
            const int d0 = (tid & 7) * 64;
            for (int j = 0; j < 64; ++j) {
                const int d = d0 + j;
                kT[d * 32 + p] = ok ? (f16)krow[d] : (f16)0.f;
            }
        }
        __syncthreads();

        union { v16h v; unsigned u[8]; } A, B;
        #pragma unroll
        for (int v = 0; v < 8; ++v)
            A.u[v] = wexpu[(nl * NPAD + kc * 32 + kpairA16(v, hi)) >> 1];
        const unsigned* kTu = (const unsigned*)kT;
        #pragma unroll
        for (int t = 0; t < 4; ++t) {
            const int dim = w * 64 + t * 16 + nl;
            #pragma unroll
            for (int v = 0; v < 8; ++v)
                B.u[v] = kTu[(dim * 32 + kpairB16(v, hi)) >> 1];
            acc[t].v = __builtin_amdgcn_wmma_f32_16x16x32_f16(false, A.v, false, B.v,
                                                              (short)0, acc[t].v, false, false);
        }
    }

    #pragma unroll
    for (int t = 0; t < 4; ++t) {
        const int dim = w * 64 + t * 16 + nl;
        #pragma unroll
        for (int r = 0; r < 8; ++r) {
            const int m = 8 * hi + r;
            const int head = ht * 16 + m;
            out[((size_t)b * HEADS + head) * DIM + dim] = acc[t].f[r] * rcpsum[m];
        }
    }
}

extern "C" void kernel_launch(void* const* d_in, const int* in_sizes, int n_in,
                              void* d_out, int out_size, void* d_ws, size_t ws_size,
                              hipStream_t stream) {
    (void)in_sizes; (void)n_in; (void)out_size; (void)ws_size;
    const float* q          = (const float*)d_in[0];
    const float* cosb       = (const float*)d_in[1];
    const float* sinb       = (const float*)d_in[2];
    const float* kv_cache   = (const float*)d_in[3];
    const float* q_idx      = (const float*)d_in[4];
    const float* k_idx      = (const float*)d_in[5];
    const float* attn_sink  = (const float*)d_in[6];
    const int*   btab       = (const int*)d_in[7];
    const int*   seq_lens   = (const int*)d_in[8];
    float* out = (float*)d_out;

    char* ws = (char*)d_ws;
    unsigned* qq   = (unsigned*)(ws + OFF_QQ);
    float* qs      = (float*)(ws + OFF_QS);
    f16* qr        = (f16*)(ws + OFF_QR);
    float* iscore  = (float*)(ws + OFF_ISC);
    int* keep_idx  = (int*)(ws + OFF_KIDX);
    int* keep_cnt  = (int*)(ws + OFF_KCNT);

    k_quant_q<<<BATCH, 128, 0, stream>>>(q_idx, qq, qs);
    k_rope<<<dim3(HEADS, BATCH), 64, 0, stream>>>(q, cosb, sinb, qr);
    k_iscore<<<dim3(MBPB, BATCH), 128, 0, stream>>>(k_idx, btab, seq_lens, qq, qs, iscore);
    k_topk<<<BATCH, 256, 0, stream>>>(iscore, keep_idx, keep_cnt);
    k_attn<<<dim3(HEADS / 16, BATCH), 256, 0, stream>>>(qr, kv_cache, btab, keep_idx,
                                                        keep_cnt, attn_sink, out);
}
